// XerxesSparseMoeBlock_48653389529594
// MI455X (gfx1250) — compile-verified, matrix operand order checked by reading, not run on hardware
//
#include <hip/hip_runtime.h>
#include <math.h>

// ---------------- problem constants (B=4,S=2048 -> T=8192) ----------------
#define TT 8192   // B*S tokens
#define HD 1024   // hidden
#define ID 2048   // intermediate
#define NE 8      // experts

typedef __attribute__((ext_vector_type(16))) __bf16 v16bf;
typedef __attribute__((ext_vector_type(8)))  __bf16 v8bf;
typedef __attribute__((ext_vector_type(8)))  float  v8f;

// ---------------- helpers ----------------
__device__ __forceinline__ unsigned short f32_bf16(float f) {
  union { float f; unsigned u; } v; v.f = f;
  unsigned u = v.u;
  u += 0x7FFFu + ((u >> 16) & 1u);   // round-to-nearest-even
  return (unsigned short)(u >> 16);
}

__device__ __forceinline__ float gelu_tanh(float x) {
  float x3 = x * x * x;
  return 0.5f * x * (1.0f + tanhf(0.7978845608028654f * (x + 0.044715f * x3)));
}

// Issue one per-lane async global->LDS 128-bit copy (gfx1250, tracked by ASYNCcnt).
__device__ __forceinline__ void async_copy_b128(const void* gptr, void* lptr) {
  unsigned int lds_addr = (unsigned int)(uintptr_t)lptr;      // LDS offset = addr[31:0]
  unsigned long long gaddr = (unsigned long long)(uintptr_t)gptr;
  asm volatile("global_load_async_to_lds_b128 %0, %1, off"
               :: "v"(lds_addr), "v"(gaddr) : "memory");
}

__device__ __forceinline__ void wait_async_all() {
  asm volatile("s_wait_asynccnt 0x0" ::: "memory");
}

// Load one 16x32 bf16 fragment (A layout) from a row-major [rows][32] LDS tile.
// Lane L: row = r0 + (L%16); K elems = {kb..kb+7, kb+16..kb+23}, kb = 8*(L/16).
// B fragments use the same pattern because B is stored N-major/K-contiguous.
__device__ __forceinline__ v16bf lds_frag(const unsigned short* base, int r0, int lane) {
  const unsigned short* p = base + (r0 + (lane & 15)) * 32 + ((lane >> 4) << 3);
  union { v16bf v; v8bf h[2]; } u;
  u.h[0] = *(const v8bf*)(p);
  u.h[1] = *(const v8bf*)(p + 16);
  return u.v;
}

// ---------------- fp32 -> bf16 conversion (vectorized, no transpose) ----------------
__global__ __launch_bounds__(256)
void cvt_bf16_kernel(const float* __restrict__ src, unsigned short* __restrict__ dst, int n4) {
  int i = blockIdx.x * blockDim.x + threadIdx.x;
  if (i >= n4) return;
  float4 f = ((const float4*)src)[i];
  unsigned short o[4];
  o[0] = f32_bf16(f.x); o[1] = f32_bf16(f.y);
  o[2] = f32_bf16(f.z); o[3] = f32_bf16(f.w);
  ((uint2*)dst)[i] = *(const uint2*)o;
}

// ---------------- fp32 [R][C] -> bf16 [C][R] tiled transpose (per expert via z) ----------------
__global__ __launch_bounds__(256)
void cvt_transpose_kernel(const float* __restrict__ src0, unsigned short* __restrict__ dst0,
                          int R, int C) {
  const float* src = src0 + (size_t)blockIdx.z * R * C;
  unsigned short* dst = dst0 + (size_t)blockIdx.z * R * C;
  __shared__ unsigned short t[64][72];   // padded to dodge bank conflicts
  const int r0 = blockIdx.y * 64, c0 = blockIdx.x * 64;
  const int tid = threadIdx.x;
#pragma unroll
  for (int p = 0; p < 4; ++p) {
    int lin = p * 1024 + tid * 4;
    int r = lin >> 6, c = lin & 63;
    float4 f = *(const float4*)&src[(size_t)(r0 + r) * C + c0 + c];
    t[r][c + 0] = f32_bf16(f.x); t[r][c + 1] = f32_bf16(f.y);
    t[r][c + 2] = f32_bf16(f.z); t[r][c + 3] = f32_bf16(f.w);
  }
  __syncthreads();
#pragma unroll
  for (int p = 0; p < 4; ++p) {
    int lin = p * 1024 + tid * 4;
    int cr = lin >> 6, rr = lin & 63;
    unsigned short o[4] = { t[rr + 0][cr], t[rr + 1][cr], t[rr + 2][cr], t[rr + 3][cr] };
    *(uint2*)&dst[(size_t)(c0 + cr) * R + r0 + rr] = *(const uint2*)o;
  }
}

// ---------------- router: logits + top-2 softmax -> combine [TT][NE] ----------------
__global__ __launch_bounds__(256)
void router_kernel(const float* __restrict__ X, const float* __restrict__ GW,
                   float* __restrict__ logits_out, float* __restrict__ combine) {
  int gwid = (blockIdx.x * blockDim.x + threadIdx.x) >> 5;   // one wave32 per token
  int lane = threadIdx.x & 31;
  if (gwid >= TT) return;

  const float* x = X + (size_t)gwid * HD;
  float acc[NE];
#pragma unroll
  for (int e = 0; e < NE; ++e) acc[e] = 0.0f;

  for (int h = lane; h < HD; h += 32) {
    float xv = x[h];
    const float4* g = (const float4*)&GW[h * NE];
    float4 g0 = g[0], g1 = g[1];
    acc[0] += xv * g0.x; acc[1] += xv * g0.y; acc[2] += xv * g0.z; acc[3] += xv * g0.w;
    acc[4] += xv * g1.x; acc[5] += xv * g1.y; acc[6] += xv * g1.z; acc[7] += xv * g1.w;
  }
#pragma unroll
  for (int off = 16; off >= 1; off >>= 1)
#pragma unroll
    for (int e = 0; e < NE; ++e) acc[e] += __shfl_down(acc[e], off, 32);

  if (lane == 0) {
#pragma unroll
    for (int e = 0; e < NE; ++e) logits_out[(size_t)gwid * NE + e] = acc[e];
    int i1 = 0;
#pragma unroll
    for (int e = 1; e < NE; ++e) if (acc[e] > acc[i1]) i1 = e;
    int i2 = (i1 == 0) ? 1 : 0;
#pragma unroll
    for (int e = 0; e < NE; ++e) if (e != i1 && acc[e] > acc[i2]) i2 = e;
    float p1 = 1.0f / (1.0f + expf(acc[i2] - acc[i1]));   // softmax over {l1,l2}
#pragma unroll
    for (int e = 0; e < NE; ++e) combine[(size_t)gwid * NE + e] = 0.0f;
    combine[(size_t)gwid * NE + i1] = p1;
    combine[(size_t)gwid * NE + i2] = 1.0f - p1;
  }
}

// ---------------- fused gate+up GEMM: act = gelu(X@Wg) * (X@Wu), bf16 out ----------------
// Weights pre-transposed: Wgt/Wut are [ID][HD] (N-major, K-contiguous).
// Tile 128x128, BK=32, 8 waves; wave = 32x64 (2x4 WMMA 16x16 tiles) for BOTH GEMMs.
// Double-buffered LDS: tile k+1 DMA (ASYNCcnt) overlaps tile-k WMMAs.
__global__ __launch_bounds__(256)
void moe_gateup_kernel(const unsigned short* __restrict__ Xb,    // [TT][HD] bf16
                       const unsigned short* __restrict__ Wgt,   // [ID][HD] bf16 (one expert)
                       const unsigned short* __restrict__ Wut,   // [ID][HD] bf16 (one expert)
                       unsigned short* __restrict__ act)         // [TT][ID] bf16
{
  __shared__ unsigned short sA [2][128 * 32];   // token-rows x K
  __shared__ unsigned short sBg[2][128 * 32];   // n-rows x K
  __shared__ unsigned short sBu[2][128 * 32];

  const int tid  = threadIdx.x;
  const int row0 = blockIdx.y * 128;   // token rows
  const int col0 = blockIdx.x * 128;   // I cols
  const int lane = tid & 31;
  const int wid  = tid >> 5;
  const int wm   = (wid & 3) * 32;     // wave row offset
  const int wn   = (wid >> 2) * 64;    // wave col offset

  auto stage = [&](int k0, int buf) {
#pragma unroll
    for (int p = 0; p < 2; ++p) {
      int lin = (p * 256 + tid) * 8;           // in shorts
      int r = lin >> 5, kk = lin & 31;         // r: tile row (tokens for A, n for B)
      async_copy_b128(&Xb [(size_t)(row0 + r) * HD + k0 + kk], &sA [buf][r * 32 + kk]);
      async_copy_b128(&Wgt[(size_t)(col0 + r) * HD + k0 + kk], &sBg[buf][r * 32 + kk]);
      async_copy_b128(&Wut[(size_t)(col0 + r) * HD + k0 + kk], &sBu[buf][r * 32 + kk]);
    }
  };

  const v8f vz = {0.f, 0.f, 0.f, 0.f, 0.f, 0.f, 0.f, 0.f};
  v8f accg[2][4], accu[2][4];
#pragma unroll
  for (int a = 0; a < 2; ++a)
#pragma unroll
    for (int b = 0; b < 4; ++b) { accg[a][b] = vz; accu[a][b] = vz; }

  const int nk = HD / 32;
  stage(0, 0);
  wait_async_all();
  __syncthreads();

  for (int ks = 0; ks < nk; ++ks) {
    const int buf = ks & 1;
    if (ks + 1 < nk) stage((ks + 1) * 32, buf ^ 1);   // DMA next tile during compute

    v16bf af[2];
    af[0] = lds_frag(sA[buf], wm,      lane);
    af[1] = lds_frag(sA[buf], wm + 16, lane);
#pragma unroll
    for (int tn = 0; tn < 4; ++tn) {
      v16bf bg = lds_frag(sBg[buf], wn + tn * 16, lane);
      v16bf bu = lds_frag(sBu[buf], wn + tn * 16, lane);
#pragma unroll
      for (int tm = 0; tm < 2; ++tm) {
        accg[tm][tn] = __builtin_amdgcn_wmma_f32_16x16x32_bf16(
            false, af[tm], false, bg, (short)0, accg[tm][tn], false, false);
        accu[tm][tn] = __builtin_amdgcn_wmma_f32_16x16x32_bf16(
            false, af[tm], false, bu, (short)0, accu[tm][tn], false, false);
      }
    }
    if (ks + 1 < nk) {
      wait_async_all();    // next tile resident
      __syncthreads();     // all readers of buf done; cross-wave visibility of tile k+1
    }
  }

  // epilogue: act = gelu(g) * u  (C layout: lane -> col=lane%16, vgpr r -> row r+8*(lane/16))
  const int rbase = row0 + wm + ((lane >> 4) << 3);
  const int cbase = col0 + wn + (lane & 15);
#pragma unroll
  for (int tm = 0; tm < 2; ++tm)
#pragma unroll
    for (int tn = 0; tn < 4; ++tn)
#pragma unroll
      for (int r = 0; r < 8; ++r) {
        int row = rbase + tm * 16 + r;
        int col = cbase + tn * 16;
        float g = accg[tm][tn][r];
        float u = accu[tm][tn][r];
        act[(size_t)row * ID + col] = f32_bf16(gelu_tanh(g) * u);
      }
}

// ---------------- down GEMM: out (+)= combine[t,e] * (act @ Wd) ----------------
// Wdt pre-transposed: [HD][ID] (N-major, K-contiguous). Double-buffered LDS.
__global__ __launch_bounds__(256)
void moe_down_kernel(const unsigned short* __restrict__ act,  // [TT][ID] bf16
                     const unsigned short* __restrict__ Wdt,  // [HD][ID] bf16 (one expert)
                     const float* __restrict__ combine,       // [TT][NE]
                     float* __restrict__ out,                 // [TT][HD] fp32
                     int e, int first)
{
  __shared__ unsigned short sA[2][128 * 32];
  __shared__ unsigned short sB[2][128 * 32];

  const int tid  = threadIdx.x;
  const int row0 = blockIdx.y * 128;   // token rows
  const int col0 = blockIdx.x * 128;   // H cols
  const int lane = tid & 31;
  const int wid  = tid >> 5;
  const int wm   = (wid & 3) * 32;
  const int wn   = (wid >> 2) * 64;

  auto stage = [&](int k0, int buf) {
#pragma unroll
    for (int p = 0; p < 2; ++p) {
      int lin = (p * 256 + tid) * 8;
      int r = lin >> 5, kk = lin & 31;
      async_copy_b128(&act[(size_t)(row0 + r) * ID + k0 + kk], &sA[buf][r * 32 + kk]);
      async_copy_b128(&Wdt[(size_t)(col0 + r) * ID + k0 + kk], &sB[buf][r * 32 + kk]);
    }
  };

  const v8f vz = {0.f, 0.f, 0.f, 0.f, 0.f, 0.f, 0.f, 0.f};
  v8f acc[2][4];
#pragma unroll
  for (int a = 0; a < 2; ++a)
#pragma unroll
    for (int b = 0; b < 4; ++b) acc[a][b] = vz;

  const int nk = ID / 32;
  stage(0, 0);
  wait_async_all();
  __syncthreads();

  for (int ks = 0; ks < nk; ++ks) {
    const int buf = ks & 1;
    if (ks + 1 < nk) stage((ks + 1) * 32, buf ^ 1);

    v16bf af[2];
    af[0] = lds_frag(sA[buf], wm,      lane);
    af[1] = lds_frag(sA[buf], wm + 16, lane);
#pragma unroll
    for (int tn = 0; tn < 4; ++tn) {
      v16bf bf = lds_frag(sB[buf], wn + tn * 16, lane);
#pragma unroll
      for (int tm = 0; tm < 2; ++tm)
        acc[tm][tn] = __builtin_amdgcn_wmma_f32_16x16x32_bf16(
            false, af[tm], false, bf, (short)0, acc[tm][tn], false, false);
    }
    if (ks + 1 < nk) {
      wait_async_all();
      __syncthreads();
    }
  }

  const int rbase = row0 + wm + ((lane >> 4) << 3);
  const int cbase = col0 + wn + (lane & 15);
#pragma unroll
  for (int tm = 0; tm < 2; ++tm)
#pragma unroll
    for (int tn = 0; tn < 4; ++tn)
#pragma unroll
      for (int r = 0; r < 8; ++r) {
        int row = rbase + tm * 16 + r;
        int col = cbase + tn * 16;
        float w = combine[(size_t)row * NE + e];
        float v = w * acc[tm][tn][r];
        float* o = &out[(size_t)row * HD + col];
        if (first) *o = v; else *o += v;   // experts serialized on stream; unique (row,col) per block
      }
}

// ---------------- host-side launch ----------------
extern "C" void kernel_launch(void* const* d_in, const int* in_sizes, int n_in,
                              void* d_out, int out_size, void* d_ws, size_t ws_size,
                              hipStream_t stream) {
  const float* X  = (const float*)d_in[0];   // [B,S,H]
  const float* GW = (const float*)d_in[1];   // [H,E]
  const float* Wg = (const float*)d_in[2];   // [E,H,I]
  const float* Wu = (const float*)d_in[3];   // [E,H,I]
  const float* Wd = (const float*)d_in[4];   // [E,I,H]

  float* out    = (float*)d_out;                 // [B,S,H] fp32
  float* logits = out + (size_t)TT * HD;         // [B,S,E] fp32

  char* ws = (char*)d_ws;
  unsigned short* Xb   = (unsigned short*)ws;  ws += (size_t)TT * HD * 2;
  unsigned short* Wgt  = (unsigned short*)ws;  ws += (size_t)NE * HD * ID * 2;  // [E][ID][HD]
  unsigned short* Wut  = (unsigned short*)ws;  ws += (size_t)NE * HD * ID * 2;  // [E][ID][HD]
  unsigned short* Wdt  = (unsigned short*)ws;  ws += (size_t)NE * ID * HD * 2;  // [E][HD][ID]
  unsigned short* Act  = (unsigned short*)ws;  ws += (size_t)TT * ID * 2;
  float*          cmb  = (float*)ws;           // [TT][NE]

  // fp32 -> bf16 staging (X direct; weights transposed to N-major/K-contiguous)
  {
    int n4x = TT * HD / 4;
    cvt_bf16_kernel<<<(n4x + 255) / 256, 256, 0, stream>>>(X, Xb, n4x);
    dim3 ggu(ID / 64, HD / 64, NE);   // [HD][ID] -> [ID][HD]
    cvt_transpose_kernel<<<ggu, 256, 0, stream>>>(Wg, Wgt, HD, ID);
    cvt_transpose_kernel<<<ggu, 256, 0, stream>>>(Wu, Wut, HD, ID);
    dim3 gdn(HD / 64, ID / 64, NE);   // [ID][HD] -> [HD][ID]
    cvt_transpose_kernel<<<gdn, 256, 0, stream>>>(Wd, Wdt, ID, HD);
  }

  // router (fp32) -> logits output + combine weights
  router_kernel<<<TT / 8, 256, 0, stream>>>(X, GW, logits, cmb);

  // dense per-expert MLP with weighted accumulation (faithful to reference)
  for (int e = 0; e < NE; ++e) {
    dim3 g1(ID / 128, TT / 128);
    moe_gateup_kernel<<<g1, 256, 0, stream>>>(
        Xb, Wgt + (size_t)e * HD * ID, Wut + (size_t)e * HD * ID, Act);
    dim3 g2(HD / 128, TT / 128);
    moe_down_kernel<<<g2, 256, 0, stream>>>(
        Act, Wdt + (size_t)e * ID * HD, cmb, out, e, e == 0 ? 1 : 0);
  }
}
